// NonLocal2d_23227183137170
// MI455X (gfx1250) — matrix operand order repaired and checked by reading, not compiled, not run on hardware
//
#include <hip/hip_runtime.h>
#include <hip/hip_bf16.h>

// Problem constants (from reference)
#define BDIM 4
#define CDIM 256
#define HIDD 128
#define NPIX 4096

typedef __attribute__((ext_vector_type(16))) __bf16 v16bf;
typedef __attribute__((ext_vector_type(8)))  float  v8f;

// GCC-style int4 vector: matches the async-to-LDS builtin's parameter type
// (diagnostic showed: 'int __attribute__((__vector_size__(4*sizeof(int)))) __device__ *')
typedef int v4i __attribute__((vector_size(16)));
typedef __attribute__((address_space(1))) v4i* gptr_v4i;
typedef __attribute__((address_space(3))) v4i* lptr_v4i;

#if __has_builtin(__builtin_amdgcn_global_load_async_to_lds_b128) && \
    __has_builtin(__builtin_amdgcn_s_wait_asynccnt)
#define HAVE_ASYNC_LDS 1
#else
#define HAVE_ASYNC_LDS 0
#endif

// ---- WMMA fragment index maps (wave32, v_wmma_f32_16x16x32_bf16) ----
// A (16x32 bf16): lane holds row M=lane&15; K = (lane>>4)*16 + e (same order as B)
//   -> per lane, one contiguous 16-element run [half*16, half*16+16), loaded as two 16B halves.
// B (32x16 bf16): lane holds col N=lane&15; K = (lane>>4)*16 + e
//   -> per lane, one contiguous 16-element run when source is K-contiguous.
// C/D (16x16 f32): lane holds col N=lane&15; register r holds row M = r + 8*(lane>>4).

__device__ __forceinline__ v8f wmma_bf16(v16bf a, v16bf b, v8f c) {
  return __builtin_amdgcn_wmma_f32_16x16x32_bf16(
      false, a, false, b, (short)0, c, false, false);
}

__device__ __forceinline__ __bf16 f2bf(float f) { return (__bf16)f; }

union FragU { uint4 u[2]; v16bf v; };

// A fragment from bf16 memory: base = &M_row_start[k0]  (row = lane&15 applied
// by caller), two 16B runs at +half*16 and +half*16+8.
__device__ __forceinline__ v16bf load_a_bf16(const __bf16* base, int half) {
  FragU t;
  t.u[0] = *(const uint4*)(base + half * 16);
  t.u[1] = *(const uint4*)(base + half * 16 + 8);
  return t.v;
}
// A fragment from f32 memory (weights), converting to bf16.
__device__ __forceinline__ v16bf load_a_f32(const float* base, int half) {
  const float4* p0 = (const float4*)(base + half * 16);
  const float4* p1 = (const float4*)(base + half * 16 + 8);
  float4 r0 = p0[0], r1 = p0[1], r2 = p1[0], r3 = p1[1];
  const float t[16] = {r0.x, r0.y, r0.z, r0.w, r1.x, r1.y, r1.z, r1.w,
                       r2.x, r2.y, r2.z, r2.w, r3.x, r3.y, r3.z, r3.w};
  v16bf a;
#pragma unroll
  for (int e = 0; e < 16; ++e) a[e] = f2bf(t[e]);
  return a;
}
// B fragment: one 32B contiguous run (caller passes fully-resolved address
// including the half*16 K offset).
__device__ __forceinline__ v16bf load_b_contig(const __bf16* base) {
  return *(const v16bf*)base;
}

// ============================================================================
// Kernel 1: fused QKV 1x1-conv projections (GEMM, K=256) with WMMA bf16.
// Workspace layouts (bf16):
//   Qn, Kt : [B][N][HID]  (pixel-major: a 64-key block of K is contiguous 16KB)
//   Vt     : [B][C][N]    (channel-major: matches the PV B-fragment needs and
//                          the C/D store pattern writes contiguous 32B runs)
// Grid: (N/64, 8, B); block = 128 threads = 4 waves, 16 rows x 64 pixels each.
// ============================================================================
__global__ __launch_bounds__(128) void qkv_proj_kernel(
    const float* __restrict__ x,
    const float* __restrict__ Wq, const float* __restrict__ bq,
    const float* __restrict__ Wk, const float* __restrict__ bk,
    const float* __restrict__ Wv, const float* __restrict__ bv,
    __bf16* __restrict__ Qn, __bf16* __restrict__ Kt, __bf16* __restrict__ Vt) {
  const int tid  = threadIdx.x;
  const int lane = tid & 31, wave = tid >> 5;
  const int l15  = lane & 15, half = lane >> 4;
  const int n0   = blockIdx.x * 64;
  const int zg   = blockIdx.y;
  const int b    = blockIdx.z;

  const int o_base = zg * 64 + wave * 16;
  const float* Wsel; const float* bsel; __bf16* dst; int o_loc, ns, os;
  if (o_base < 128) {          // Q -> [N][HID]
    Wsel = Wq; bsel = bq; dst = Qn + (size_t)b * NPIX * HIDD;
    o_loc = o_base; ns = HIDD; os = 1;
  } else if (o_base < 256) {   // K -> [N][HID]
    Wsel = Wk; bsel = bk; dst = Kt + (size_t)b * NPIX * HIDD;
    o_loc = o_base - 128; ns = HIDD; os = 1;
  } else {                     // V -> [C][N]
    Wsel = Wv; bsel = bv; dst = Vt + (size_t)b * CDIM * NPIX;
    o_loc = o_base - 256; ns = 1; os = NPIX;
  }
  const float* xb = x + (size_t)b * CDIM * NPIX;

  v8f acc[4];
#pragma unroll
  for (int t = 0; t < 4; ++t)
#pragma unroll
    for (int i = 0; i < 8; ++i) acc[t][i] = 0.f;

  for (int kc = 0; kc < 8; ++kc) {             // K loop over 256 channels
    const int c0 = kc * 32;
    v16bf a = load_a_f32(Wsel + (size_t)(o_loc + l15) * CDIM + c0, half);
#pragma unroll
    for (int t = 0; t < 4; ++t) {
      const int col = n0 + t * 16 + l15;
      v16bf bb;                                // x is [C][N]: K strided, lanes coalesced
#pragma unroll
      for (int e = 0; e < 16; ++e)
        bb[e] = f2bf(xb[(size_t)(c0 + (half << 4) + e) * NPIX + col]);
      acc[t] = wmma_bf16(a, bb, acc[t]);
    }
  }

#pragma unroll
  for (int t = 0; t < 4; ++t) {
    const int n = n0 + t * 16 + l15;
#pragma unroll
    for (int r = 0; r < 8; ++r) {
      const int orow = o_loc + r + 8 * half;
      dst[(size_t)n * ns + (size_t)orow * os] = f2bf(acc[t][r] + bsel[orow]);
    }
  }
}

// ============================================================================
// Kernel 2: flash attention. Block = 256 threads = 8 waves; each wave owns 16
// query rows (128 rows per block). Per 64-key step: K tile [64][128] and V tile
// [256][64] (K-contiguous for B-fragments) staged in LDS via async-to-LDS
// (ASYNCcnt) when available, S = Q K^T (16 wmma), online softmax, P staged
// through per-wave LDS patch, O += P V (32 wmma). All fragment loads are b128.
// ============================================================================
__global__ __launch_bounds__(256) void attention_kernel(
    const float* __restrict__ x,
    const __bf16* __restrict__ Qn, const __bf16* __restrict__ Kt,
    const __bf16* __restrict__ Vt, float* __restrict__ out) {
  __shared__ __align__(32) __bf16 Kblk[64 * HIDD];    // [key][hid]   16 KB
  __shared__ __align__(32) __bf16 VblkT[CDIM * 64];   // [c][key]     32 KB
  __shared__ __align__(32) __bf16 Pst[8 * 16 * 64];   // per-wave 2KB 16 KB

  const int tid  = threadIdx.x;
  const int lane = tid & 31, wave = tid >> 5;
  const int l15  = lane & 15, half = lane >> 4;
  const int b    = blockIdx.y;
  const int i0   = blockIdx.x * 128 + wave * 16;

  const __bf16* Qb = Qn + (size_t)b * NPIX * HIDD;
  const __bf16* Kb = Kt + (size_t)b * NPIX * HIDD;
  const __bf16* Vb = Vt + (size_t)b * CDIM * NPIX;

  // Q fragments, resident for the whole pass (4 frags = 32 VGPRs)
  v16bf qa[4];
#pragma unroll
  for (int kc = 0; kc < 4; ++kc)
    qa[kc] = load_a_bf16(Qb + (size_t)(i0 + l15) * HIDD + kc * 32, half);

  v8f o_acc[16];
#pragma unroll
  for (int ct = 0; ct < 16; ++ct)
#pragma unroll
    for (int i = 0; i < 8; ++i) o_acc[ct][i] = 0.f;

  float m_run[8], l_run[8];
#pragma unroll
  for (int r = 0; r < 8; ++r) { m_run[r] = -__builtin_inff(); l_run[r] = 0.f; }

  for (int jb = 0; jb < NPIX / 64; ++jb) {
    const int j0 = jb * 64;
    __syncthreads();                        // previous iteration's LDS reads done
    {
      // K tile: contiguous 16KB (1024 uint4 over 256 threads)
      const uint4* sk = (const uint4*)(Kb + (size_t)j0 * HIDD);
      uint4* dk = (uint4*)Kblk;
      // V tile: row c of [C][N] -> VblkT[c][0..63]; one 128B row per thread
      const uint4* sv = (const uint4*)(Vb + (size_t)tid * NPIX + j0);
      uint4* dv = (uint4*)(VblkT + tid * 64);
#if HAVE_ASYNC_LDS
#pragma unroll
      for (int it = 0; it < 4; ++it) {
        const int idx = tid + it * 256;
        __builtin_amdgcn_global_load_async_to_lds_b128(
            (gptr_v4i)(sk + idx), (lptr_v4i)(dk + idx), 0, 0);
      }
#pragma unroll
      for (int u = 0; u < 8; ++u)
        __builtin_amdgcn_global_load_async_to_lds_b128(
            (gptr_v4i)(sv + u), (lptr_v4i)(dv + u), 0, 0);
      __builtin_amdgcn_s_wait_asynccnt(0);
#else
#pragma unroll
      for (int it = 0; it < 4; ++it) {
        const int idx = tid + it * 256;
        dk[idx] = sk[idx];
      }
#pragma unroll
      for (int u = 0; u < 8; ++u) dv[u] = sv[u];
#endif
    }
    if (jb + 1 < NPIX / 64) {               // global_prefetch_b8 for next tiles
      __builtin_prefetch(Kb + (size_t)(j0 + 64) * HIDD + tid * 32, 0, 1);
      __builtin_prefetch(Vb + (size_t)tid * NPIX + j0 + 64, 0, 1);
    }
    __syncthreads();

    // ---- S = Q * K_blk^T (16x64, K-dim 128): 16 wmma, b128 fragment loads ----
    v8f s[4];
#pragma unroll
    for (int t = 0; t < 4; ++t) {
#pragma unroll
      for (int i = 0; i < 8; ++i) s[t][i] = 0.f;
#pragma unroll
      for (int kc = 0; kc < 4; ++kc) {
        v16bf bb = load_b_contig(Kblk + (t * 16 + l15) * HIDD + kc * 32 + half * 16);
        s[t] = wmma_bf16(qa[kc], bb, s[t]);
      }
    }

    // ---- online softmax; rows live in 16-lane half-groups ----
#pragma unroll
    for (int r = 0; r < 8; ++r) {
      float mx = fmaxf(fmaxf(s[0][r], s[1][r]), fmaxf(s[2][r], s[3][r]));
#pragma unroll
      for (int d = 1; d < 16; d <<= 1) mx = fmaxf(mx, __shfl_xor(mx, d, 32));
      const float mnew = fmaxf(m_run[r], mx);
      const float corr = __expf(m_run[r] - mnew);
      m_run[r] = mnew;
      l_run[r] *= corr;
#pragma unroll
      for (int ct = 0; ct < 16; ++ct) o_acc[ct][r] *= corr;
      float psum = 0.f;
#pragma unroll
      for (int t = 0; t < 4; ++t) {
        const float p = __expf(s[t][r] - mnew);
        psum += p;
        Pst[wave * 1024 + (r + 8 * half) * 64 + t * 16 + l15] = f2bf(p);
      }
#pragma unroll
      for (int d = 1; d < 16; d <<= 1) psum += __shfl_xor(psum, d, 32);
      l_run[r] += psum;
    }
    // same-wave DS store->load ordering before reloading P as A-fragments
    asm volatile("s_wait_dscnt 0" ::: "memory");

    // ---- O += P * V_blk (16x64 x 64x256): 32 wmma ----
    v16bf pa[2];
#pragma unroll
    for (int kc = 0; kc < 2; ++kc)
      pa[kc] = load_a_bf16(Pst + wave * 1024 + l15 * 64 + kc * 32, half);
#pragma unroll
    for (int ct = 0; ct < 16; ++ct) {
#pragma unroll
      for (int kc = 0; kc < 2; ++kc) {
        // B[K=j][col=c] from VblkT[c][j]: contiguous in K
        v16bf bb = load_b_contig(VblkT + (ct * 16 + l15) * 64 + kc * 32 + half * 16);
        o_acc[ct] = wmma_bf16(pa[kc], bb, o_acc[ct]);
      }
    }
  }

  // ---- epilogue: normalize, residual add, store f32 [B,C,N] ----
#pragma unroll
  for (int r = 0; r < 8; ++r) {
    const float inv = 1.0f / l_run[r];
    const int i = i0 + r + 8 * half;
#pragma unroll
    for (int ct = 0; ct < 16; ++ct) {
      const int c = ct * 16 + l15;
      const size_t idx = ((size_t)b * CDIM + c) * NPIX + i;
      out[idx] = x[idx] + o_acc[ct][r] * inv;
    }
  }
}

// ============================================================================
extern "C" void kernel_launch(void* const* d_in, const int* in_sizes, int n_in,
                              void* d_out, int out_size, void* d_ws, size_t ws_size,
                              hipStream_t stream) {
  const float* x  = (const float*)d_in[0];
  const float* Wq = (const float*)d_in[1];
  const float* bq = (const float*)d_in[2];
  const float* Wk = (const float*)d_in[3];
  const float* bk = (const float*)d_in[4];
  const float* Wv = (const float*)d_in[5];
  const float* bv = (const float*)d_in[6];
  float* out = (float*)d_out;

  // bf16 workspace: Qn, Kt [B][N][HID]; Vt [B][C][N]  (16 MB total)
  __bf16* Qn = (__bf16*)d_ws;
  __bf16* Kt = Qn + (size_t)BDIM * NPIX * HIDD;
  __bf16* Vt = Kt + (size_t)BDIM * NPIX * HIDD;

  qkv_proj_kernel<<<dim3(NPIX / 64, 8, BDIM), 128, 0, stream>>>(
      x, Wq, bq, Wk, bk, Wv, bv, Qn, Kt, Vt);
  attention_kernel<<<dim3(NPIX / 128, BDIM), 256, 0, stream>>>(
      x, Qn, Kt, Vt, out);
}
